// RelModel_67190468378870
// MI455X (gfx1250) — matrix operand order repaired and checked by reading, not compiled, not run on hardware
//
#include <hip/hip_runtime.h>
#include <hip/hip_bf16.h>
#include <math.h>

// ---------------------------------------------------------------------------
// Types for CDNA5 WMMA (wave32): bf16 A/B fragments (v16bf) + f32 accum (v8f)
// ---------------------------------------------------------------------------
typedef __bf16        v16bf __attribute__((ext_vector_type(16)));
typedef float         v8f   __attribute__((ext_vector_type(8)));
typedef unsigned int  u32x4 __attribute__((ext_vector_type(4)));

union Frag32B { v16bf v; u32x4 q[2]; };

static __device__ __forceinline__ v8f vzero8() {
  v8f z; for (int i = 0; i < 8; ++i) z[i] = 0.0f; return z;
}

// float -> bf16 bits, round-to-nearest-even
static __device__ __forceinline__ unsigned short f2bf(float x) {
  union { float f; unsigned int u; } a; a.f = x;
  unsigned int u = a.u;
  u += 0x7FFFu + ((u >> 16) & 1u);
  return (unsigned short)(u >> 16);
}

static __device__ __forceinline__ float sigmoidf(float x) {
  return 1.0f / (1.0f + __expf(-x));
}

// A fragment (16x32 bf16) from row-major bf16 matrix, per ISA layout:
//   lanes 0-15 : m=lane,    dwords0-3 K=[kc*32+0..7],  dwords4-7 K=[kc*32+16..23]
//   lanes16-31 : m=lane-16, dwords0-3 K=[kc*32+8..15], dwords4-7 K=[kc*32+24..31]
static __device__ __forceinline__ v16bf load_a_frag(const unsigned short* A, int ldk,
                                                    int mrow, int kc, int lane) {
  int m  = mrow + (lane & 15);
  int hi = (lane >> 4) & 1;
  const unsigned short* row = A + (size_t)m * ldk + kc * 32;
  Frag32B f;
  f.q[0] = *reinterpret_cast<const u32x4*>(row + hi * 8);
  f.q[1] = *reinterpret_cast<const u32x4*>(row + 16 + hi * 8);
  return f.v;
}

// B fragment (32x16 bf16) from pre-packed layout:
//   packed[((nt*Kchunks + kc)*32 + lane)*16 + e] holds
//   W[k = kc*32 + (lane>>4)*8 + (e&7) + (e>>3)*16][n = nt*16 + (lane&15)]
static __device__ __forceinline__ v16bf load_b_frag(const unsigned short* Bp,
                                                    int Kchunks, int nt, int kc, int lane) {
  const unsigned short* p = Bp + ((((size_t)nt * Kchunks + kc) * 32 + lane) * 16);
  Frag32B f;
  f.q[0] = reinterpret_cast<const u32x4*>(p)[0];
  f.q[1] = reinterpret_cast<const u32x4*>(p)[1];
  return f.v;
}

static __device__ __forceinline__ v8f wmma_bf16(v16bf a, v16bf b, v8f c) {
  return __builtin_amdgcn_wmma_f32_16x16x32_bf16(false, a, false, b, (short)0, c, false, false);
}

// ---------------------------------------------------------------------------
// Problem constants
// ---------------------------------------------------------------------------
#define NUM_IM   32
#define ROIS     128
#define NROWS    4096       // NUM_IM*ROIS
#define NCLS     151
#define EMBED    200
#define OBJ_DIM  2048
#define HALF_H   256
#define H4       1024
#define D_OBJ    2376       // 2048+200+128
#define D_OBJ_P  2400       // padded to mult of 32
#define D_EDGE   2760       // 200+512+2048
#define D_EDGE_P 2784
#define D_DEC0   2376       // feats part of dec_w
#define D_DEC1   512        // enc part of dec_w
#define NDEC_P   160        // 151 padded
#define OUTW     663        // 151 + 512

// ---------------------------------------------------------------------------
// Weight packers: f32 -> bf16 WMMA-B fragment layout
// ---------------------------------------------------------------------------
// src layout (2, Kvalid, 1024); dst: N=2048 (dir-major), Ntiles=128
__global__ void pack2dir_kernel(const float* __restrict__ W, int Kvalid, int Kpad,
                                unsigned short* __restrict__ dst, long total) {
  long i = (long)blockIdx.x * blockDim.x + threadIdx.x;
  if (i >= total) return;
  int e    = (int)(i & 15);
  int lane = (int)((i >> 4) & 31);
  long rest = i >> 9;
  int Kchunks = Kpad >> 5;
  int kc = (int)(rest % Kchunks);
  int nt = (int)(rest / Kchunks);
  int n  = nt * 16 + (lane & 15);
  int k  = kc * 32 + ((lane >> 4) << 3) + (e & 7) + ((e >> 3) << 4);
  int dir = n >> 10, c = n & 1023;
  float v = (k < Kvalid) ? W[((size_t)dir * Kvalid + k) * 1024 + c] : 0.0f;
  dst[i] = f2bf(v);
}

// src layout (rows x ld) with row offset; dst Ntiles tiles of 16 cols
__global__ void pack1_kernel(const float* __restrict__ W, int ld, int rowOff,
                             int Kvalid, int Kpad, int Nvalid, int Ntiles,
                             unsigned short* __restrict__ dst, long total) {
  long i = (long)blockIdx.x * blockDim.x + threadIdx.x;
  if (i >= total) return;
  int e    = (int)(i & 15);
  int lane = (int)((i >> 4) & 31);
  long rest = i >> 9;
  int Kchunks = Kpad >> 5;
  int kc = (int)(rest % Kchunks);
  int nt = (int)(rest / Kchunks);
  int n  = nt * 16 + (lane & 15);
  int k  = kc * 32 + ((lane >> 4) << 3) + (e & 7) + ((e >> 3) << 4);
  float v = (k < Kvalid && n < Nvalid) ? W[(size_t)(rowOff + k) * ld + n] : 0.0f;
  dst[i] = f2bf(v);
}

// ---------------------------------------------------------------------------
// Fused softmax + obj_embed + BN/pos-MLP + feats assembly (bf16) + conf
// one block (256 thr) per row
// ---------------------------------------------------------------------------
__global__ void features_kernel(const float* __restrict__ logits,
                                const float* __restrict__ fmaps,
                                const float* __restrict__ boxes,
                                const float* __restrict__ embW,
                                const float* __restrict__ bnG, const float* __restrict__ bnB,
                                const float* __restrict__ bnM, const float* __restrict__ bnV,
                                const float* __restrict__ posW, const float* __restrict__ posB,
                                unsigned short* __restrict__ feats,
                                float* __restrict__ conf) {
  int r = blockIdx.x, t = threadIdx.x;
  __shared__ float probs[NCLS];
  __shared__ float red[256];

  // softmax
  float mx = -INFINITY;
  for (int c = t; c < NCLS; c += 256) mx = fmaxf(mx, logits[(size_t)r * NCLS + c]);
  red[t] = mx; __syncthreads();
  for (int st = 128; st > 0; st >>= 1) { if (t < st) red[t] = fmaxf(red[t], red[t + st]); __syncthreads(); }
  float m = red[0]; __syncthreads();

  float part = 0.0f;
  for (int c = t; c < NCLS; c += 256) {
    float e = __expf(logits[(size_t)r * NCLS + c] - m);
    probs[c] = e; part += e;
  }
  red[t] = part; __syncthreads();
  for (int st = 128; st > 0; st >>= 1) { if (t < st) red[t] += red[t + st]; __syncthreads(); }
  float rS = 1.0f / red[0]; __syncthreads();
  for (int c = t; c < NCLS; c += 256) probs[c] *= rS;
  __syncthreads();

  // conf = max probs[1:]
  float cm = -INFINITY;
  for (int c = 1 + t; c < NCLS; c += 256) cm = fmaxf(cm, probs[c]);
  red[t] = cm; __syncthreads();
  for (int st = 128; st > 0; st >>= 1) { if (t < st) red[t] = fmaxf(red[t], red[t + st]); __syncthreads(); }
  if (t == 0) conf[r] = red[0];

  unsigned short* row = feats + (size_t)r * D_OBJ_P;

  // obj_embed = probs @ embW  (151x200)
  if (t < EMBED) {
    float a = 0.0f;
    for (int k = 0; k < NCLS; ++k) a += probs[k] * embW[(size_t)k * EMBED + t];
    row[OBJ_DIM + t] = f2bf(a);
  }

  // pos = relu(bn(center_size(boxes)) @ posW + posB)
  if (t < 128) {
    float x1 = boxes[(size_t)r * 4 + 0], y1 = boxes[(size_t)r * 4 + 1];
    float x2 = boxes[(size_t)r * 4 + 2], y2 = boxes[(size_t)r * 4 + 3];
    float cs[4] = { (x1 + x2) * 0.5f, (y1 + y2) * 0.5f, x2 - x1, y2 - y1 };
    float a = posB[t];
    for (int k = 0; k < 4; ++k) {
      float csn = (cs[k] - bnM[k]) * rsqrtf(bnV[k] + 1e-5f) * bnG[k] + bnB[k];
      a += csn * posW[k * 128 + t];
    }
    row[OBJ_DIM + EMBED + t] = f2bf(fmaxf(a, 0.0f));
  }

  // fmaps part + zero pad
  for (int c = t; c < OBJ_DIM; c += 256) row[c] = f2bf(fmaps[(size_t)r * OBJ_DIM + c]);
  if (t < D_OBJ_P - D_OBJ) row[D_OBJ + t] = 0;
}

// ---------------------------------------------------------------------------
// Per-image stable descending bitonic sort of conf (mimics argsort(-conf))
// ---------------------------------------------------------------------------
__global__ void sort_kernel(const float* __restrict__ conf, int* __restrict__ perm,
                            int* __restrict__ inv) {
  int b = blockIdx.x, t = threadIdx.x;
  __shared__ float key[ROIS];
  __shared__ int   idx[ROIS];
  key[t] = conf[b * ROIS + t]; idx[t] = t;
  __syncthreads();
  for (int k = 2; k <= ROIS; k <<= 1) {
    for (int j = k >> 1; j > 0; j >>= 1) {
      int ixj = t ^ j;
      if (ixj > t) {
        float ka = key[t], kb = key[ixj];
        int ia = idx[t], ib = idx[ixj];
        bool aFirst = (ka > kb) || (ka == kb && ia < ib);   // desc, stable
        bool up = ((t & k) == 0);
        if (up ? !aFirst : aFirst) { key[t] = kb; key[ixj] = ka; idx[t] = ib; idx[ixj] = ia; }
      }
      __syncthreads();
    }
  }
  perm[b * ROIS + t] = idx[t];
  inv[b * ROIS + idx[t]] = t;
}

// ---------------------------------------------------------------------------
// Gather rows by perm (bf16, stride multiple of 8)
// ---------------------------------------------------------------------------
__global__ void gather_rows_kernel(const unsigned short* __restrict__ src,
                                   const int* __restrict__ perm,
                                   unsigned short* __restrict__ dst, int stride) {
  int s = blockIdx.x;
  int b = s >> 7;
  int p = b * ROIS + perm[s];
  const u32x4* sp = (const u32x4*)(src + (size_t)p * stride);
  u32x4*       dp = (u32x4*)(dst + (size_t)s * stride);
  int n = stride >> 3;
  for (int c = threadIdx.x; c < n; c += 256) dp[c] = sp[c];
}

// ---------------------------------------------------------------------------
// Register-blocked bf16 WMMA GEMM for the big projections (N multiple of 256):
//   C[4096, N] (f32) = A[4096, Kpad] @ Bpacked + bias
// block = 256 thr = 8 waves as 2(M) x 4(N); wave tile = 32x64 (2 Mt x 4 Nt).
// Each A fragment feeds 4 WMMAs, each B fragment feeds 2 -> 1.5 b128 loads/WMMA.
// ---------------------------------------------------------------------------
__global__ void gemm_bf16_blocked_kernel(const unsigned short* __restrict__ A, int lda,
                                         const unsigned short* __restrict__ Bp,
                                         float* __restrict__ C, int ldc,
                                         const float* __restrict__ bias,
                                         int Kchunks) {
  int lane = threadIdx.x & 31, wv = threadIdx.x >> 5;
  int mt0 = blockIdx.x * 4 + (wv >> 2) * 2;   // two consecutive 16-row tiles
  int nt0 = blockIdx.y * 16 + (wv & 3) * 4;   // four consecutive 16-col tiles
  v8f acc[2][4];
  for (int i = 0; i < 2; ++i)
    for (int j = 0; j < 4; ++j) acc[i][j] = vzero8();

  for (int kc = 0; kc < Kchunks; ++kc) {
    v16bf a0 = load_a_frag(A, lda, mt0 * 16, kc, lane);
    v16bf a1 = load_a_frag(A, lda, mt0 * 16 + 16, kc, lane);
    for (int j = 0; j < 4; ++j) {
      v16bf b = load_b_frag(Bp, Kchunks, nt0 + j, kc, lane);
      acc[0][j] = wmma_bf16(a0, b, acc[0][j]);
      acc[1][j] = wmma_bf16(a1, b, acc[1][j]);
    }
  }

  for (int i = 0; i < 2; ++i) {
    int r0 = (mt0 + i) * 16 + ((lane < 16) ? 0 : 8);
    for (int j = 0; j < 4; ++j) {
      int n = (nt0 + j) * 16 + (lane & 15);
      float bv = (bias != nullptr) ? bias[n] : 0.0f;
      for (int v = 0; v < 8; ++v)
        C[(size_t)(r0 + v) * ldc + n] = bv + acc[i][j][v];
    }
  }
}

// ---------------------------------------------------------------------------
// Generic (guarded) bf16 WMMA GEMM for small N (decoder, N tiles may be ragged)
// block = 256 thr = 8 waves (2 M-tiles x 4 N-tiles), wave = one 16x16 C tile
// ---------------------------------------------------------------------------
__global__ void gemm_bf16_kernel(const unsigned short* __restrict__ A, int lda,
                                 const unsigned short* __restrict__ Bp,
                                 float* __restrict__ C, int ldc,
                                 const float* __restrict__ bias,
                                 int Kchunks, int Ntiles, int Nvalid, int accumulate) {
  int lane = threadIdx.x & 31, wv = threadIdx.x >> 5;
  int mt = blockIdx.x * 2 + (wv >> 2);
  int nt = blockIdx.y * 4 + (wv & 3);
  if (nt >= Ntiles) return;                 // wave-uniform
  int mrow = mt * 16;
  v8f acc = vzero8();
  for (int kc = 0; kc < Kchunks; ++kc) {
    v16bf a = load_a_frag(A, lda, mrow, kc, lane);
    v16bf b = load_b_frag(Bp, Kchunks, nt, kc, lane);
    acc = wmma_bf16(a, b, acc);
  }
  int n  = nt * 16 + (lane & 15);
  int r0 = mrow + ((lane < 16) ? 0 : 8);
  float bv = (bias != nullptr && n < Nvalid) ? bias[n] : 0.0f;
  for (int v = 0; v < 8; ++v) {
    size_t idx = (size_t)(r0 + v) * ldc + n;
    float cur = accumulate ? C[idx] : bv;
    C[idx] = cur + acc[v];
  }
}

// ---------------------------------------------------------------------------
// BiLSTM recurrence (one direction, 16 images) — per-step (16x256)@(256x1024)
// WMMA GEMM with h in LDS (bf16) and c in registers, gates fused in-register.
// grid = (2 dirs, 2 image halves), block = 512 thr = 16 waves.
// pre rows are sorted order (img*128 + t), 2048 cols = [dir0 gates | dir1 gates]
// ---------------------------------------------------------------------------
__global__ void lstm_recur_kernel(const float* __restrict__ pre,
                                  const unsigned short* __restrict__ WhhP,  // packed, Ntiles=128, Kchunks=8
                                  unsigned short* __restrict__ hsB,         // [4096][512] sorted bf16
                                  float* __restrict__ hsF) {                // nullable f32 copy
  const int dir = blockIdx.x;
  const int ig  = blockIdx.y;                    // images [ig*16, ig*16+16)
  const int lane = threadIdx.x & 31, wv = threadIdx.x >> 5;   // 16 waves: h cols [wv*16, wv*16+16)
  __shared__ unsigned short h_lds[16 * HALF_H];  // 16 imgs x 256 cols bf16

  for (int i = threadIdx.x; i < 16 * HALF_H; i += 512) h_lds[i] = 0;
  v8f c_reg = vzero8();
  __syncthreads();

  const int Kc = 8;                              // 256 / 32
  const int ncolBase = dir * 1024 + wv * 16 + (lane & 15);
  const int imgL0 = (lane < 16) ? 0 : 8;

  for (int s = 0; s < ROIS; ++s) {
    int u = dir ? (ROIS - 1 - s) : s;

    // init accumulators with pre-activations (pre already includes bias)
    v8f acc[4];
    for (int g = 0; g < 4; ++g) {
      v8f a;
      int ncol = ncolBase + g * HALF_H;
      for (int v = 0; v < 8; ++v) {
        int img = ig * 16 + imgL0 + v;
        a[v] = pre[((size_t)img * ROIS + u) * 2048 + ncol];
      }
      acc[g] = a;
    }

    // h @ whh via WMMA
    for (int kc = 0; kc < Kc; ++kc) {
      v16bf af = load_a_frag(h_lds, HALF_H, 0, kc, lane);
      for (int g = 0; g < 4; ++g) {
        int nt = dir * 64 + g * 16 + wv;
        v16bf bf = load_b_frag(WhhP, Kc, nt, kc, lane);
        acc[g] = wmma_bf16(af, bf, acc[g]);
      }
    }
    __syncthreads();   // all waves done reading h_lds

    // fused gates: C-tile layout maps i/f/g/o lane-for-lane
    int colh = wv * 16 + (lane & 15);
    for (int v = 0; v < 8; ++v) {
      float iv = sigmoidf(acc[0][v]);
      float fv = sigmoidf(acc[1][v]);
      float gv = tanhf(acc[2][v]);
      float ov = sigmoidf(acc[3][v]);
      float cn = fv * c_reg[v] + iv * gv;
      c_reg[v] = cn;
      float hn = ov * tanhf(cn);
      int imgL = imgL0 + v;
      unsigned short hb = f2bf(hn);
      h_lds[imgL * HALF_H + colh] = hb;
      size_t row = (size_t)(ig * 16 + imgL) * ROIS + u;
      hsB[row * 512 + dir * HALF_H + colh] = hb;
      if (hsF) hsF[row * 512 + dir * HALF_H + colh] = hn;
    }
    __syncthreads();   // new h visible before next step
  }
}

// ---------------------------------------------------------------------------
// Unsort sorted hs (bf16) back to original row order
// ---------------------------------------------------------------------------
__global__ void unsort_bf16_kernel(const unsigned short* __restrict__ hsB,
                                   const int* __restrict__ inv,
                                   unsigned short* __restrict__ enc) {
  int r = blockIdx.x;
  int b = r >> 7;
  size_t s = (size_t)b * ROIS + inv[r];
  for (int c = threadIdx.x; c < 512; c += 256)
    enc[(size_t)r * 512 + c] = hsB[s * 512 + c];
}

// ---------------------------------------------------------------------------
// Argmax over dists[:,1:151] (first-max tiebreak) + copy dists -> out[:, :151]
// ---------------------------------------------------------------------------
__global__ void argmax_kernel(const float* __restrict__ dists, int* __restrict__ preds,
                              float* __restrict__ out) {
  int r = blockIdx.x, t = threadIdx.x;
  __shared__ float bv[256];
  __shared__ int   bi[256];
  float best = -INFINITY; int bidx = 0x7fffffff;
  for (int c = 1 + t; c < NCLS; c += 256) {
    float v = dists[(size_t)r * NDEC_P + c];
    if (v > best || (v == best && c < bidx)) { best = v; bidx = c; }
  }
  bv[t] = best; bi[t] = bidx; __syncthreads();
  for (int st = 128; st > 0; st >>= 1) {
    if (t < st) {
      if (bv[t + st] > bv[t] || (bv[t + st] == bv[t] && bi[t + st] < bi[t])) {
        bv[t] = bv[t + st]; bi[t] = bi[t + st];
      }
    }
    __syncthreads();
  }
  if (t == 0) preds[r] = bi[0];   // == argmax(dists[:,1:]) + 1
  for (int c = t; c < NCLS; c += 256)
    out[(size_t)r * OUTW + c] = dists[(size_t)r * NDEC_P + c];
}

// ---------------------------------------------------------------------------
// Build sorted edge_in rows: [embed2[pred], enc, fmaps] (bf16, padded)
// ---------------------------------------------------------------------------
__global__ void gather_edge_kernel(const float* __restrict__ emb2,
                                   const int* __restrict__ preds,
                                   const unsigned short* __restrict__ encB,
                                   const float* __restrict__ fmaps,
                                   const int* __restrict__ perm,
                                   unsigned short* __restrict__ xs) {
  int s = blockIdx.x;
  int b = s >> 7;
  int p = b * ROIS + perm[s];
  unsigned short* row = xs + (size_t)s * D_EDGE_P;
  int t = threadIdx.x;
  int cls = preds[p];
  for (int c = t; c < EMBED; c += 256)   row[c]       = f2bf(emb2[(size_t)cls * EMBED + c]);
  for (int c = t; c < 512; c += 256)     row[200 + c] = encB[(size_t)p * 512 + c];
  for (int c = t; c < OBJ_DIM; c += 256) row[712 + c] = f2bf(fmaps[(size_t)p * OBJ_DIM + c]);
  if (t < D_EDGE_P - D_EDGE) row[D_EDGE + t] = 0;
}

// ---------------------------------------------------------------------------
// Final: out[:, 151:663] = unsorted edge hs (f32)
// ---------------------------------------------------------------------------
__global__ void finalize_kernel(const float* __restrict__ hsF, const int* __restrict__ inv,
                                float* __restrict__ out) {
  int r = blockIdx.x;
  int b = r >> 7;
  size_t s = (size_t)b * ROIS + inv[r];
  for (int c = threadIdx.x; c < 512; c += 256)
    out[(size_t)r * OUTW + NCLS + c] = hsF[s * 512 + c];
}

// ---------------------------------------------------------------------------
// Host launcher
// ---------------------------------------------------------------------------
extern "C" void kernel_launch(void* const* d_in, const int* in_sizes, int n_in,
                              void* d_out, int out_size, void* d_ws, size_t ws_size,
                              hipStream_t stream) {
  (void)in_sizes; (void)n_in; (void)out_size; (void)ws_size;

  const float* obj_fmaps   = (const float*)d_in[0];
  const float* obj_logits  = (const float*)d_in[1];
  const float* boxes       = (const float*)d_in[2];
  /* d_in[3] im_inds unused (row->image is r/128) */
  const float* obj_embed_w  = (const float*)d_in[4];
  const float* obj_embed2_w = (const float*)d_in[5];
  const float* bn_gamma = (const float*)d_in[6];
  const float* bn_beta  = (const float*)d_in[7];
  const float* bn_mean  = (const float*)d_in[8];
  const float* bn_var   = (const float*)d_in[9];
  const float* pos_w    = (const float*)d_in[10];
  const float* pos_b    = (const float*)d_in[11];
  const float* obj_wih0 = (const float*)d_in[12];
  const float* obj_whh0 = (const float*)d_in[13];
  const float* obj_b0   = (const float*)d_in[14];   // (2,1024) contiguous = 2048 bias
  const float* obj_wih1 = (const float*)d_in[15];
  const float* obj_whh1 = (const float*)d_in[16];
  const float* obj_b1   = (const float*)d_in[17];
  const float* edge_wih0 = (const float*)d_in[18];
  const float* edge_whh0 = (const float*)d_in[19];
  const float* edge_b0   = (const float*)d_in[20];
  const float* edge_wih1 = (const float*)d_in[21];
  const float* edge_whh1 = (const float*)d_in[22];
  const float* edge_b1   = (const float*)d_in[23];
  const float* dec_w     = (const float*)d_in[24];
  const float* dec_b     = (const float*)d_in[25];
  float* out = (float*)d_out;

  // ---- workspace layout (all sizes 256B-multiples) ----
  char* ws = (char*)d_ws;
  size_t off = 0;
  auto alloc = [&](size_t bytes) { char* p = ws + off; off += (bytes + 255) & ~(size_t)255; return p; };
  unsigned short* FEATS = (unsigned short*)alloc((size_t)NROWS * D_OBJ_P * 2);
  unsigned short* XS    = (unsigned short*)alloc((size_t)NROWS * D_EDGE_P * 2);
  float*          PRE   = (float*)alloc((size_t)NROWS * 2048 * 4);
  unsigned short* HSB   = (unsigned short*)alloc((size_t)NROWS * 512 * 2);
  float*          HSF   = (float*)alloc((size_t)NROWS * 512 * 4);
  unsigned short* ENCB  = (unsigned short*)alloc((size_t)NROWS * 512 * 2);
  float*          DISTS = (float*)alloc((size_t)NROWS * NDEC_P * 4);
  float*          CONF  = (float*)alloc((size_t)NROWS * 4);
  int*            PERM  = (int*)alloc((size_t)NROWS * 4);
  int*            INV   = (int*)alloc((size_t)NROWS * 4);
  int*            PREDS = (int*)alloc((size_t)NROWS * 4);
  unsigned short* W_OIH0 = (unsigned short*)alloc((size_t)D_OBJ_P * 2048 * 2);
  unsigned short* W_OHH0 = (unsigned short*)alloc((size_t)256 * 2048 * 2);
  unsigned short* W_OIH1 = (unsigned short*)alloc((size_t)512 * 2048 * 2);
  unsigned short* W_OHH1 = (unsigned short*)alloc((size_t)256 * 2048 * 2);
  unsigned short* W_EIH0 = (unsigned short*)alloc((size_t)D_EDGE_P * 2048 * 2);
  unsigned short* W_EHH0 = (unsigned short*)alloc((size_t)256 * 2048 * 2);
  unsigned short* W_EIH1 = (unsigned short*)alloc((size_t)512 * 2048 * 2);
  unsigned short* W_EHH1 = (unsigned short*)alloc((size_t)256 * 2048 * 2);
  unsigned short* W_DEC0 = (unsigned short*)alloc((size_t)D_OBJ_P * NDEC_P * 2);
  unsigned short* W_DEC1 = (unsigned short*)alloc((size_t)512 * NDEC_P * 2);

  auto pack2 = [&](const float* src, int Kv, int Kp, unsigned short* dst) {
    long total = (long)Kp * 2048;
    pack2dir_kernel<<<(unsigned)((total + 255) / 256), 256, 0, stream>>>(src, Kv, Kp, dst, total);
  };
  // ---- 1) weight packing ----
  pack2(obj_wih0, D_OBJ, D_OBJ_P, W_OIH0);
  pack2(obj_whh0, 256, 256, W_OHH0);
  pack2(obj_wih1, 512, 512, W_OIH1);
  pack2(obj_whh1, 256, 256, W_OHH1);
  pack2(edge_wih0, D_EDGE, D_EDGE_P, W_EIH0);
  pack2(edge_whh0, 256, 256, W_EHH0);
  pack2(edge_wih1, 512, 512, W_EIH1);
  pack2(edge_whh1, 256, 256, W_EHH1);
  {
    long t0 = (long)D_OBJ_P * NDEC_P;
    pack1_kernel<<<(unsigned)((t0 + 255) / 256), 256, 0, stream>>>(dec_w, NCLS, 0, D_DEC0, D_OBJ_P, NCLS, NDEC_P / 16, W_DEC0, t0);
    long t1 = (long)512 * NDEC_P;
    pack1_kernel<<<(unsigned)((t1 + 255) / 256), 256, 0, stream>>>(dec_w, NCLS, D_DEC0, D_DEC1, 512, NCLS, NDEC_P / 16, W_DEC1, t1);
  }

  // ---- 2) features + conf ----
  features_kernel<<<NROWS, 256, 0, stream>>>(obj_logits, obj_fmaps, boxes, obj_embed_w,
                                             bn_gamma, bn_beta, bn_mean, bn_var,
                                             pos_w, pos_b, FEATS, CONF);
  // ---- 3) per-image sort ----
  sort_kernel<<<NUM_IM, ROIS, 0, stream>>>(CONF, PERM, INV);
  // ---- 4) gather sorted obj inputs ----
  gather_rows_kernel<<<NROWS, 256, 0, stream>>>(FEATS, PERM, XS, D_OBJ_P);

  // ---- 5) obj BiLSTM layer 0 ----
  gemm_bf16_blocked_kernel<<<dim3(64, 8), 256, 0, stream>>>(XS, D_OBJ_P, W_OIH0, PRE, 2048,
                                                            obj_b0, D_OBJ_P / 32);
  lstm_recur_kernel<<<dim3(2, 2), 512, 0, stream>>>(PRE, W_OHH0, HSB, nullptr);
  // ---- 6) obj BiLSTM layer 1 ----
  gemm_bf16_blocked_kernel<<<dim3(64, 8), 256, 0, stream>>>(HSB, 512, W_OIH1, PRE, 2048,
                                                            obj_b1, 512 / 32);
  lstm_recur_kernel<<<dim3(2, 2), 512, 0, stream>>>(PRE, W_OHH1, HSB, nullptr);
  // ---- 7) unsort enc ----
  unsort_bf16_kernel<<<NROWS, 256, 0, stream>>>(HSB, INV, ENCB);

  // ---- 8) decoder: dists = [feats|enc] @ dec_w + dec_b (split-K, accumulate) ----
  gemm_bf16_kernel<<<dim3(128, 3), 256, 0, stream>>>(FEATS, D_OBJ_P, W_DEC0, DISTS, NDEC_P,
                                                     dec_b, D_OBJ_P / 32, NDEC_P / 16, NCLS, 0);
  gemm_bf16_kernel<<<dim3(128, 3), 256, 0, stream>>>(ENCB, 512, W_DEC1, DISTS, NDEC_P,
                                                     nullptr, 512 / 32, NDEC_P / 16, NCLS, 1);
  // ---- 9) argmax + copy dists into out ----
  argmax_kernel<<<NROWS, 256, 0, stream>>>(DISTS, PREDS, out);

  // ---- 10) edge inputs (sorted) ----
  gather_edge_kernel<<<NROWS, 256, 0, stream>>>(obj_embed2_w, PREDS, ENCB, obj_fmaps, PERM, XS);

  // ---- 11) edge BiLSTM layer 0 ----
  gemm_bf16_blocked_kernel<<<dim3(64, 8), 256, 0, stream>>>(XS, D_EDGE_P, W_EIH0, PRE, 2048,
                                                            edge_b0, D_EDGE_P / 32);
  lstm_recur_kernel<<<dim3(2, 2), 512, 0, stream>>>(PRE, W_EHH0, HSB, nullptr);
  // ---- 12) edge BiLSTM layer 1 (writes f32 too) ----
  gemm_bf16_blocked_kernel<<<dim3(64, 8), 256, 0, stream>>>(HSB, 512, W_EIH1, PRE, 2048,
                                                            edge_b1, 512 / 32);
  lstm_recur_kernel<<<dim3(2, 2), 512, 0, stream>>>(PRE, W_EHH1, HSB, HSF);

  // ---- 13) final unsort into output ----
  finalize_kernel<<<NROWS, 256, 0, stream>>>(HSF, INV, out);
}
